// MultiEmbedding_4466765987946
// MI455X (gfx1250) — matrix operand order repaired and checked by reading, not compiled, not run on hardware
//
#include <hip/hip_runtime.h>
#include <hip/hip_bf16.h>
#include <math.h>

// Problem dims (fixed by the reference)
#define BN   8
#define CN   256
#define HN   64
#define WN   64
#define PN   4096          // H*W
#define NHEAD 4
#define HDIMN 64           // head dim
#define K3   512           // concat channels for 3x3 conv

#define LDT      40                  // LDS tile row stride (bf16): 80B, 16B-aligned
#define AROWS    128                 // pixel rows per workgroup tile
#define ATILE_E  (AROWS * LDT)       // 5120 bf16
#define WTILE_E  (64 * LDT)          // 2560 bf16
#define ATILE_B  (ATILE_E * 2)       // 10240 B
#define WTILE_B  (WTILE_E * 2)       // 5120 B
#define SMEM_B   33280               // max(2*ATILE_B + 2*WTILE_B = 30720, 128*65*4 = 33280)

typedef __attribute__((ext_vector_type(16))) __bf16 bf16x16;
typedef __attribute__((ext_vector_type(8)))  __bf16 bf16x8;
typedef __attribute__((ext_vector_type(8)))  float  f32x8;

// ---------------------------------------------------------------------------
// CDNA5 async global->LDS copy (ASYNCcnt DMA). 16B per call.
// ---------------------------------------------------------------------------
__device__ __forceinline__ void async_b128(unsigned lds_off, const void* g) {
  asm volatile("global_load_async_to_lds_b128 %0, %1, off"
               :: "v"(lds_off), "v"((unsigned long long)(uintptr_t)g)
               : "memory");
}
__device__ __forceinline__ void wait_async0() {
  asm volatile("s_wait_asynccnt 0x0" ::: "memory");
}

// ---------------------------------------------------------------------------
// WMMA 16x16x32 bf16 fragments (wave32, ISA 05_wmma §7.12.2); both operands
// per-lane k = 8h+s (s<8), 16+8h+(s-8) (s>=8) -> 2x ds_load_b128 per fragment.
// ---------------------------------------------------------------------------
__device__ __forceinline__ bf16x16 frag_a(const __bf16* tile, int lane) {
  const int m = lane & 15, kh = lane >> 4;
  const __bf16* p = tile + m * LDT + kh * 8;
  bf16x8 lo = *(const bf16x8*)(p);
  bf16x8 hi = *(const bf16x8*)(p + 16);
  return __builtin_shufflevector(lo, hi, 0,1,2,3,4,5,6,7,8,9,10,11,12,13,14,15);
}
__device__ __forceinline__ bf16x16 frag_b(const __bf16* tile, int lane) {
  const int n = lane & 15, kh = lane >> 4;
  const __bf16* p = tile + n * LDT + kh * 8;
  bf16x8 lo = *(const bf16x8*)(p);
  bf16x8 hi = *(const bf16x8*)(p + 16);
  return __builtin_shufflevector(lo, hi, 0,1,2,3,4,5,6,7,8,9,10,11,12,13,14,15);
}
__device__ __forceinline__ f32x8 bwmma(bf16x16 a, bf16x16 b, f32x8 c) {
  return __builtin_amdgcn_wmma_f32_16x16x32_bf16(false, a, false, b,
                                                 (short)0, c, false, false);
}

// ---------------------------------------------------------------------------
// Prep kernels
// ---------------------------------------------------------------------------
__global__ void transpose_cvt(const float* __restrict__ in, __bf16* __restrict__ out) {
  __shared__ __bf16 sh[32][33];
  const int p0 = blockIdx.x * 32, c0 = blockIdx.y * 32, b = blockIdx.z;
  const float* ib = in + (size_t)b * CN * PN;
  __bf16* ob = out + (size_t)b * PN * CN;
  const int tx = threadIdx.x, ty = threadIdx.y;
#pragma unroll
  for (int j = 0; j < 4; ++j)
    sh[ty + j * 8][tx] = (__bf16)ib[(size_t)(c0 + ty + j * 8) * PN + p0 + tx];
  __syncthreads();
#pragma unroll
  for (int j = 0; j < 4; ++j)
    ob[(size_t)(p0 + ty + j * 8) * CN + c0 + tx] = sh[tx][ty + j * 8];
}

__global__ void cvt_f32_bf16(const float* __restrict__ in, __bf16* __restrict__ out, int n) {
  int i = blockIdx.x * blockDim.x + threadIdx.x;
  if (i < n) out[i] = (__bf16)in[i];
}

__global__ void repack_w3(const float* __restrict__ w3, __bf16* __restrict__ w3p) {
  int i = blockIdx.x * blockDim.x + threadIdx.x;   // over 256*512*9
  if (i >= CN * K3 * 9) return;
  int co  = i / (K3 * 9);
  int rem = i % (K3 * 9);
  int ci  = rem / 9;
  int tap = rem % 9;
  w3p[(size_t)tap * CN * K3 + (size_t)co * K3 + ci] = (__bf16)w3[i];
}

// ---------------------------------------------------------------------------
// Pixel-major 1x1-conv GEMM:  D[p][co] = sum_ci XT[p][ci] * W[co][ci]
// Workgroup tile 128 px x 64 co; wave tile 32x32 (4 WMMAs / 8 ds_load_b128).
// Async double-buffered K-chunks of 32.
// EPI 0: bf16 out = acc + bias[co]   (q/k/v)
// EPI 1: bf16 out = BN(acc)          (wW + BatchNorm)
// EPI 2: f32 NCHW out via LDS transpose (final 1x1 + bias)
// ---------------------------------------------------------------------------
template <int EPI>
__global__ void gemm_pm(const __bf16* __restrict__ XT,  // [B][PN][K]
                        const __bf16* __restrict__ Wt,  // [256][K]
                        float* __restrict__ outF,       // EPI2: [B][256][PN]
                        __bf16* __restrict__ outB,      // EPI<2: [B][PN][256]
                        const float* __restrict__ p0,
                        const float* __restrict__ p1,
                        const float* __restrict__ p2,
                        const float* __restrict__ p3,
                        int K) {
  __shared__ __align__(16) char smem[SMEM_B];
  __bf16* As = (__bf16*)smem;                       // 2 x 128x40
  __bf16* Ws = (__bf16*)(smem + 2 * ATILE_B);       // 2 x 64x40

  const int b     = blockIdx.z;
  const int nbase = blockIdx.y * 64;    // co tile
  const int pbase = blockIdx.x * 128;   // pixel tile

  const int t    = threadIdx.x;
  const int lane = t & 31;
  const int wv   = t >> 5;
  const int msub = (wv & 3) * 32;       // pixel subtile (2x16)
  const int nsub = (wv >> 2) * 32;      // co subtile (2x16)

  const int row  = t >> 2, colg = (t & 3) * 8;   // 16B chunk coords

  const __bf16* Xr0 = XT + ((size_t)b * PN + pbase + row) * K + colg;
  const __bf16* Xr1 = Xr0 + (size_t)64 * K;
  const __bf16* Wr  = Wt + (size_t)(nbase + row) * K + colg;
  const unsigned smB  = (unsigned)(uintptr_t)smem;
  const unsigned aDst = smB + (unsigned)(row * LDT + colg) * 2u;
  const unsigned wDst = smB + 2u * ATILE_B + (unsigned)(row * LDT + colg) * 2u;

  const int nk = K / 32;
  f32x8 acc00 = {}, acc01 = {}, acc10 = {}, acc11 = {};

  auto issue = [&](int buf, int kc) {
    const unsigned ao = (unsigned)(buf * ATILE_B), wo = (unsigned)(buf * WTILE_B);
    async_b128(aDst + ao, Xr0 + kc);
    async_b128(aDst + ao + (unsigned)(64 * LDT * 2), Xr1 + kc);
    async_b128(wDst + wo, Wr + kc);
  };

  issue(0, 0);
  for (int i = 0; i < nk; ++i) {
    wait_async0();
    __syncthreads();
    if (i + 1 < nk) issue((i + 1) & 1, (i + 1) * 32);
    const __bf16* as = As + (i & 1) * ATILE_E;
    const __bf16* ws = Ws + (i & 1) * WTILE_E;
    bf16x16 a0 = frag_a(as + msub * LDT, lane);
    bf16x16 a1 = frag_a(as + (msub + 16) * LDT, lane);
    bf16x16 b0 = frag_b(ws + nsub * LDT, lane);
    bf16x16 b1 = frag_b(ws + (nsub + 16) * LDT, lane);
    acc00 = bwmma(a0, b0, acc00);
    acc01 = bwmma(a0, b1, acc01);
    acc10 = bwmma(a1, b0, acc10);
    acc11 = bwmma(a1, b1, acc11);
  }

  const int n  = lane & 15;
  const int hi = lane >> 4;

  if (EPI == 2) {
    __syncthreads();                       // tiles dead; reuse pool as D stage
    float* Dsh = (float*)smem;             // [128][65]
    const float bb0 = p0[nbase + nsub + n], bb1 = p0[nbase + nsub + n + 16];
#pragma unroll
    for (int s = 0; s < 2; ++s) {
      const f32x8 aL = s ? acc10 : acc00;
      const f32x8 aR = s ? acc11 : acc01;
#pragma unroll
      for (int v = 0; v < 8; ++v) {
        const int m = msub + s * 16 + v + 8 * hi;
        Dsh[m * 65 + nsub + n]      = aL[v] + bb0;
        Dsh[m * 65 + nsub + n + 16] = aR[v] + bb1;
      }
    }
    __syncthreads();
    const int co_l = t >> 2, pg = (t & 3) * 32;
    float* orow = outF + (size_t)b * CN * PN + (size_t)(nbase + co_l) * PN + pbase + pg;
#pragma unroll
    for (int j = 0; j < 8; ++j) {
      float4 o;
      o.x = Dsh[(pg + 4 * j + 0) * 65 + co_l];
      o.y = Dsh[(pg + 4 * j + 1) * 65 + co_l];
      o.z = Dsh[(pg + 4 * j + 2) * 65 + co_l];
      o.w = Dsh[(pg + 4 * j + 3) * 65 + co_l];
      *(float4*)(orow + 4 * j) = o;
    }
  } else {
    const int co0 = nbase + nsub + n;
    float s0 = 0.f, s1 = 0.f, o0 = 0.f, o1 = 0.f;
    if (EPI == 0) {
      o0 = p0[co0]; o1 = p0[co0 + 16]; s0 = 1.f; s1 = 1.f;
    } else {  // BatchNorm inference: y = x*inv + (beta - mean*inv)
      s0 = p0[co0]      * rsqrtf(p3[co0] + 1e-5f);
      s1 = p0[co0 + 16] * rsqrtf(p3[co0 + 16] + 1e-5f);
      o0 = p1[co0]      - p2[co0]      * s0;
      o1 = p1[co0 + 16] - p2[co0 + 16] * s1;
    }
#pragma unroll
    for (int s = 0; s < 2; ++s) {
      const f32x8 aL = s ? acc10 : acc00;
      const f32x8 aR = s ? acc11 : acc01;
#pragma unroll
      for (int v = 0; v < 8; ++v) {
        const int p = pbase + msub + s * 16 + v + 8 * hi;
        __bf16* od = outB + ((size_t)b * PN + p) * CN + co0;
        od[0]  = (__bf16)(aL[v] * s0 + o0);   // lane-consecutive co: coalesced
        od[16] = (__bf16)(aR[v] * s1 + o1);
      }
    }
  }
}

// ---------------------------------------------------------------------------
// Head-softmax attention, pixel-major (contiguous per-pixel channel rows).
// ---------------------------------------------------------------------------
__global__ void attn_softmax(const __bf16* __restrict__ q,
                             const __bf16* __restrict__ k,
                             const __bf16* __restrict__ v,
                             __bf16* __restrict__ wout) {
  const int g = blockIdx.x * blockDim.x + threadIdx.x;  // pixel over B*PN
  if (g >= BN * PN) return;
  const bf16x8* q8 = (const bf16x8*)(q + (size_t)g * CN);
  const bf16x8* k8 = (const bf16x8*)(k + (size_t)g * CN);
  const bf16x8* v8 = (const bf16x8*)(v + (size_t)g * CN);

  float att[NHEAD];
#pragma unroll
  for (int h = 0; h < NHEAD; ++h) {
    float s = 0.f;
#pragma unroll
    for (int j = 0; j < 8; ++j) {
      bf16x8 a = q8[h * 8 + j], c = k8[h * 8 + j];
#pragma unroll
      for (int e = 0; e < 8; ++e) s += (float)a[e] * (float)c[e];
    }
    att[h] = s * (1.0f / 16.0f);  // 1/sqrt(256)
  }
  float mx = fmaxf(fmaxf(att[0], att[1]), fmaxf(att[2], att[3]));
  float se = 0.f;
#pragma unroll
  for (int h = 0; h < NHEAD; ++h) { att[h] = __expf(att[h] - mx); se += att[h]; }
  const float rs = 1.0f / se;
#pragma unroll
  for (int h = 0; h < NHEAD; ++h) att[h] *= rs;

  bf16x8* w8 = (bf16x8*)(wout + (size_t)g * HDIMN);
#pragma unroll
  for (int j = 0; j < 8; ++j) {
    bf16x8 o;
#pragma unroll
    for (int e = 0; e < 8; ++e) {
      float s = 0.f;
#pragma unroll
      for (int h = 0; h < NHEAD; ++h) s += att[h] * (float)v8[h * 8 + j][e];
      o[e] = (__bf16)s;
    }
    w8[j] = o;
  }
}

// ---------------------------------------------------------------------------
// 3x3 reflect-pad conv as implicit GEMM (pixel-major), tile 128 px x 64 co,
// 144 async double-buffered K-chunks (9 taps x 512/32); ELU epilogue.
// ---------------------------------------------------------------------------
__global__ void conv3x3_elu(const __bf16* __restrict__ w3p,   // [tap][256][512]
                            const __bf16* __restrict__ keyT,  // [B][PN][256]
                            const __bf16* __restrict__ aoT,   // [B][PN][256]
                            const float* __restrict__ b3,
                            __bf16* __restrict__ out) {       // [B][PN][256]
  __shared__ __align__(16) char smem[2 * ATILE_B + 2 * WTILE_B];
  __bf16* As = (__bf16*)smem;
  __bf16* Ws = (__bf16*)(smem + 2 * ATILE_B);

  const int pbase = blockIdx.x * 128;   // two image rows
  const int nbase = blockIdx.y * 64;
  const int b     = blockIdx.z;

  const int t    = threadIdx.x;
  const int lane = t & 31;
  const int wv   = t >> 5;
  const int msub = (wv & 3) * 32;
  const int nsub = (wv >> 2) * 32;
  const int row  = t >> 2, colg = (t & 3) * 8;

  const size_t bp = (size_t)b * PN;
  const unsigned smB  = (unsigned)(uintptr_t)smem;
  const unsigned aDst = smB + (unsigned)(row * LDT + colg) * 2u;
  const unsigned wDst = smB + 2u * ATILE_B + (unsigned)(row * LDT + colg) * 2u;

  auto srcFor = [&](int rr, int dy, int dx, int kc) -> const __bf16* {
    const int iy = (pbase + rr) >> 6, ix = (pbase + rr) & 63;
    int ys = iy + dy; ys = (ys < 0) ? 1 : ((ys > HN - 1) ? HN - 2 : ys);
    int xs = ix + dx; xs = (xs < 0) ? 1 : ((xs > WN - 1) ? WN - 2 : xs);
    return ((kc < CN) ? keyT : aoT) +
           (bp + (size_t)ys * WN + xs) * CN + (kc & (CN - 1)) + colg;
  };
  auto issue = [&](int buf, int ii) {
    const int tap = ii >> 4, kc = (ii & 15) * 32;
    const int dy = tap / 3 - 1, dx = tap % 3 - 1;
    const unsigned ao = (unsigned)(buf * ATILE_B), wo = (unsigned)(buf * WTILE_B);
    async_b128(aDst + ao, srcFor(row, dy, dx, kc));
    async_b128(aDst + ao + (unsigned)(64 * LDT * 2), srcFor(row + 64, dy, dx, kc));
    async_b128(wDst + wo, w3p + ((size_t)tap * CN + nbase + row) * K3 + kc + colg);
  };

  f32x8 acc00 = {}, acc01 = {}, acc10 = {}, acc11 = {};
  issue(0, 0);
  for (int i = 0; i < 144; ++i) {
    wait_async0();
    __syncthreads();
    if (i + 1 < 144) issue((i + 1) & 1, i + 1);
    const __bf16* as = As + (i & 1) * ATILE_E;
    const __bf16* ws = Ws + (i & 1) * WTILE_E;
    bf16x16 a0 = frag_a(as + msub * LDT, lane);
    bf16x16 a1 = frag_a(as + (msub + 16) * LDT, lane);
    bf16x16 b0 = frag_b(ws + nsub * LDT, lane);
    bf16x16 b1 = frag_b(ws + (nsub + 16) * LDT, lane);
    acc00 = bwmma(a0, b0, acc00);
    acc01 = bwmma(a0, b1, acc01);
    acc10 = bwmma(a1, b0, acc10);
    acc11 = bwmma(a1, b1, acc11);
  }

  const int n  = lane & 15;
  const int hi = lane >> 4;
  const int co0 = nbase + nsub + n;
  const float bb0 = b3[co0], bb1 = b3[co0 + 16];
#pragma unroll
  for (int s = 0; s < 2; ++s) {
    const f32x8 aL = s ? acc10 : acc00;
    const f32x8 aR = s ? acc11 : acc01;
#pragma unroll
    for (int v = 0; v < 8; ++v) {
      const int p = pbase + msub + s * 16 + v + 8 * hi;
      float x0 = aL[v] + bb0;
      float x1 = aR[v] + bb1;
      x0 = (x0 > 0.f) ? x0 : (__expf(x0) - 1.f);   // ELU
      x1 = (x1 > 0.f) ? x1 : (__expf(x1) - 1.f);
      __bf16* od = out + (bp + p) * CN + co0;
      od[0]  = (__bf16)x0;
      od[16] = (__bf16)x1;
    }
  }
}

// ---------------------------------------------------------------------------
// Launcher
// ---------------------------------------------------------------------------
extern "C" void kernel_launch(void* const* d_in, const int* in_sizes, int n_in,
                              void* d_out, int out_size, void* d_ws, size_t ws_size,
                              hipStream_t stream) {
  (void)in_sizes; (void)n_in; (void)out_size; (void)ws_size;
  const float* key   = (const float*)d_in[0];
  const float* query = (const float*)d_in[1];
  const float* wq = (const float*)d_in[2];
  const float* bq = (const float*)d_in[3];
  const float* wk = (const float*)d_in[4];
  const float* bk = (const float*)d_in[5];
  const float* wv = (const float*)d_in[6];
  const float* bv = (const float*)d_in[7];
  const float* wW = (const float*)d_in[8];
  const float* gamma = (const float*)d_in[9];
  const float* beta  = (const float*)d_in[10];
  const float* mean  = (const float*)d_in[11];
  const float* var   = (const float*)d_in[12];
  const float* w3 = (const float*)d_in[13];
  const float* b3 = (const float*)d_in[14];
  const float* w1 = (const float*)d_in[15];
  const float* b1 = (const float*)d_in[16];
  float* out = (float*)d_out;

  char* wp = (char*)d_ws;
  auto carve = [&](size_t bytes) {
    char* p = wp;
    wp += (bytes + 255) & ~(size_t)255;
    return p;
  };
  const size_t fullB = (size_t)BN * CN * PN * sizeof(__bf16);   // 16.8 MB
  __bf16* keyT = (__bf16*)carve(fullB);                         // [B][P][256]
  __bf16* qryT = (__bf16*)carve(fullB);
  __bf16* wqB  = (__bf16*)carve(CN * CN * sizeof(__bf16));
  __bf16* wkB  = (__bf16*)carve(CN * CN * sizeof(__bf16));
  __bf16* wvB  = (__bf16*)carve(CN * CN * sizeof(__bf16));
  __bf16* wWB  = (__bf16*)carve(CN * HDIMN * sizeof(__bf16));
  __bf16* w1B  = (__bf16*)carve(CN * CN * sizeof(__bf16));
  __bf16* w3pB = (__bf16*)carve((size_t)9 * CN * K3 * sizeof(__bf16));
  __bf16* qB   = (__bf16*)carve(fullB);
  __bf16* kB   = (__bf16*)carve(fullB);
  __bf16* vB   = (__bf16*)carve(fullB);
  __bf16* wtB  = (__bf16*)carve((size_t)BN * PN * HDIMN * sizeof(__bf16));
  __bf16* aoB  = (__bf16*)carve(fullB);
  __bf16* fuB  = (__bf16*)carve(fullB);

  const int TB = 256;

  // --- prep ---
  transpose_cvt<<<dim3(PN / 32, CN / 32, BN), dim3(32, 8), 0, stream>>>(key,   keyT);
  transpose_cvt<<<dim3(PN / 32, CN / 32, BN), dim3(32, 8), 0, stream>>>(query, qryT);
  cvt_f32_bf16<<<(CN * CN + TB - 1) / TB, TB, 0, stream>>>(wq, wqB, CN * CN);
  cvt_f32_bf16<<<(CN * CN + TB - 1) / TB, TB, 0, stream>>>(wk, wkB, CN * CN);
  cvt_f32_bf16<<<(CN * CN + TB - 1) / TB, TB, 0, stream>>>(wv, wvB, CN * CN);
  cvt_f32_bf16<<<(CN * HDIMN + TB - 1) / TB, TB, 0, stream>>>(wW, wWB, CN * HDIMN);
  cvt_f32_bf16<<<(CN * CN + TB - 1) / TB, TB, 0, stream>>>(w1, w1B, CN * CN);
  repack_w3<<<(CN * K3 * 9 + TB - 1) / TB, TB, 0, stream>>>(w3, w3pB);

  const dim3 gGemm(PN / 128, CN / 64, BN);  // (32, 4, 8)

  // --- q/k/v projections ---
  gemm_pm<0><<<gGemm, TB, 0, stream>>>(qryT, wqB, nullptr, qB, bq, nullptr, nullptr, nullptr, CN);
  gemm_pm<0><<<gGemm, TB, 0, stream>>>(keyT, wkB, nullptr, kB, bk, nullptr, nullptr, nullptr, CN);
  gemm_pm<0><<<gGemm, TB, 0, stream>>>(keyT, wvB, nullptr, vB, bv, nullptr, nullptr, nullptr, CN);

  // --- head softmax + weighted sum ---
  attn_softmax<<<(BN * PN) / TB, TB, 0, stream>>>(qB, kB, vB, wtB);

  // --- wW (K=64) + BatchNorm ---
  gemm_pm<1><<<gGemm, TB, 0, stream>>>(wtB, wWB, nullptr, aoB, gamma, beta, mean, var, HDIMN);

  // --- 3x3 reflect-pad conv + ELU ---
  conv3x3_elu<<<gGemm, TB, 0, stream>>>(w3pB, keyT, aoB, b3, fuB);

  // --- final 1x1 + bias -> NCHW f32 d_out ---
  gemm_pm<2><<<gGemm, TB, 0, stream>>>(fuB, w1B, out, nullptr, b1, nullptr, nullptr, nullptr, CN);
}
